// RGCN_63513976373569
// MI455X (gfx1250) — matrix operand order repaired and checked by reading, not compiled, not run on hardware
//
#include <hip/hip_runtime.h>

typedef __attribute__((ext_vector_type(16))) __bf16 v16bf;
typedef __attribute__((ext_vector_type(8)))  float  v8f;

__device__ __forceinline__ unsigned short f2bf(float f) {
    union { float f; unsigned int u; } c; c.f = f;
    unsigned int u = c.u;
    u += 0x7FFFu + ((u >> 16) & 1u);   // round to nearest even
    return (unsigned short)(u >> 16);
}

// ---------------------------------------------------------------------------
// Per-relation degree counts (f32 atomics; counts < 2^24 so exact).
// ---------------------------------------------------------------------------
__global__ void rgcn_degrees(const int* __restrict__ src, const int* __restrict__ dst,
                             float* __restrict__ ds, float* __restrict__ dd,
                             int RE, int E, int N) {
    int i = blockIdx.x * blockDim.x + threadIdx.x;
    if (i >= RE) return;
    int r = i / E;
    atomicAdd(&ds[(size_t)r * N + src[i]], 1.0f);
    atomicAdd(&dd[(size_t)r * N + dst[i]], 1.0f);
}

__global__ void rgcn_norms(float* __restrict__ p, int n) {
    int i = blockIdx.x * blockDim.x + threadIdx.x;
    if (i >= n) return;
    p[i] = rsqrtf(fmaxf(p[i], 1.0f));
}

// ---------------------------------------------------------------------------
// Weights -> bf16, transposed to [r][n][k] so B fragments are contiguous.
// ---------------------------------------------------------------------------
__global__ void rgcn_wconv(const float* __restrict__ W, unsigned short* __restrict__ Wt, int total) {
    int i = blockIdx.x * blockDim.x + threadIdx.x;
    if (i >= total) return;
    int r = i >> 14;
    int n = (i >> 7) & 127;
    int k = i & 127;
    Wt[((size_t)r << 14) + (n << 7) + k] = f2bf(W[((size_t)r << 14) + (k << 7) + n]);
}

// ---------------------------------------------------------------------------
// Fused src-norm gather + scatter-add: agg[r][dst] += x[src] * rsqrt(outdeg).
// One wave per edge; lane handles a float4 (coalesced 512B row load).
// ---------------------------------------------------------------------------
__global__ void rgcn_scatter(const float* __restrict__ X, const int* __restrict__ src,
                             const int* __restrict__ dst, const float* __restrict__ s_src,
                             float* __restrict__ agg, int RE, int E, int N) {
    int gid = blockIdx.x * blockDim.x + threadIdx.x;
    int w = gid >> 5, lane = gid & 31;
    if (w >= RE) return;
    int r = w / E;
    int s = src[w], d = dst[w];
    float sc = s_src[(size_t)r * N + s];
    float4 v = ((const float4*)(X + (size_t)s * 128))[lane];
    float* o = agg + ((size_t)r * N + d) * 128 + lane * 4;
    atomicAdd(o + 0, v.x * sc);
    atomicAdd(o + 1, v.y * sc);
    atomicAdd(o + 2, v.z * sc);
    atomicAdd(o + 3, v.w * sc);
}

// ---------------------------------------------------------------------------
// Fused: dst-norm scale, sum_r (agg_r @ W_r), + sum_r b_r, optional ReLU.
// Block = 256 thr = 8 waves; 16-row M-tile, wave w owns output cols [16w,16w+16).
// K = 128 = 4 chunks of 32 per v_wmma_f32_16x16x32_bf16.
// ---------------------------------------------------------------------------
__global__ void __launch_bounds__(256)
rgcn_gemm(const float* __restrict__ agg, const float* __restrict__ s_dst,
          const unsigned short* __restrict__ Wt, const float* __restrict__ b,
          float* __restrict__ out, int N, int R, int do_relu) {
    __shared__ unsigned short tile[16][136];   // 16 x 128 bf16, padded stride
    const int tid  = threadIdx.x;
    const int lane = tid & 31;
    const int wv   = tid >> 5;                 // N-tile index 0..7
    const int m0   = blockIdx.x * 16;

    v8f c = {0.f, 0.f, 0.f, 0.f, 0.f, 0.f, 0.f, 0.f};

    const int trow = tid >> 4;                 // 0..15 (staging row)
    const int tcol = (tid & 15) * 8;           // 0..120 (staging col seg)

    for (int r = 0; r < R; ++r) {
        // Stage dst-normalized A tile as bf16 in LDS.
        {
            int node = m0 + trow;
            float4 a0 = {0.f, 0.f, 0.f, 0.f}, a1 = {0.f, 0.f, 0.f, 0.f};
            float sd = 0.f;
            if (node < N) {
                sd = s_dst[(size_t)r * N + node];
                const float* arow = agg + ((size_t)r * N + node) * 128 + tcol;
                a0 = *(const float4*)(arow);
                a1 = *(const float4*)(arow + 4);
            }
            unsigned short* t = &tile[trow][tcol];
            t[0] = f2bf(a0.x * sd); t[1] = f2bf(a0.y * sd);
            t[2] = f2bf(a0.z * sd); t[3] = f2bf(a0.w * sd);
            t[4] = f2bf(a1.x * sd); t[5] = f2bf(a1.y * sd);
            t[6] = f2bf(a1.z * sd); t[7] = f2bf(a1.w * sd);
        }
        __syncthreads();

        // This wave's 16 weight columns for relation r (bf16, [n][k] layout).
        const unsigned short* wcol =
            Wt + ((size_t)r << 14) + (size_t)(wv * 16 + (lane & 15)) * 128;
        if (r + 1 < R)
            __builtin_prefetch(wcol + (1u << 14), 0, 0);  // next relation's weights

        const int am    = lane & 15;          // A fragment: M = lane%16
        const int koff  = (lane >> 4) * 8;    // A K-halves split at lane 16
        const int bkoff = (lane >> 4) * 8;    // B K-halves split at lane 16

        #pragma unroll
        for (int kc = 0; kc < 4; ++kc) {
            v16bf Av, Bv;
            const unsigned short* ta = &tile[am][kc * 32 + koff];
            *((uint4*)(&Av) + 0) = *(const uint4*)(ta);         // K = koff..koff+7
            *((uint4*)(&Av) + 1) = *(const uint4*)(ta + 16);    // K = koff+16..+23
            const unsigned short* tb = wcol + kc * 32 + bkoff;
            *((uint4*)(&Bv) + 0) = *(const uint4*)(tb);         // K = bkoff..+7
            *((uint4*)(&Bv) + 1) = *(const uint4*)(tb + 16);    // K = bkoff+16..+23
            c = __builtin_amdgcn_wmma_f32_16x16x32_bf16(false, Av, false, Bv,
                                                        (short)0, c, false, false);
        }
        __syncthreads();
    }

    // Epilogue: bias-sum over relations, optional ReLU, store per C layout:
    // VGPR i -> row m0 + i + 8*(lane/16), col = 16*wv + lane%16.
    const int col = wv * 16 + (lane & 15);
    float bs = 0.f;
    for (int r = 0; r < R; ++r) bs += b[r * 128 + col];
    const int mh = (lane >> 4) * 8;
    #pragma unroll
    for (int i = 0; i < 8; ++i) {
        int row = m0 + mh + i;
        if (row < N) {
            float v = c[i] + bs;
            if (do_relu) v = fmaxf(v, 0.f);
            out[(size_t)row * 128 + col] = v;
        }
    }
}

// ---------------------------------------------------------------------------
extern "C" void kernel_launch(void* const* d_in, const int* in_sizes, int n_in,
                              void* d_out, int out_size, void* d_ws, size_t ws_size,
                              hipStream_t stream) {
    const float* x  = (const float*)d_in[0];
    const int*  src = (const int*)d_in[1];
    const int*  dst = (const int*)d_in[2];
    const float* W1 = (const float*)d_in[3];
    const float* b1 = (const float*)d_in[4];
    const float* W2 = (const float*)d_in[5];
    const float* b2 = (const float*)d_in[6];
    float* out = (float*)d_out;

    const int N = in_sizes[0] / 128;       // x: N*128
    const int R = in_sizes[4] / 128;       // b1: R*128
    const int E = in_sizes[1] / R;         // src: R*E
    const int RE = R * E;

    // Workspace layout
    char* ws = (char*)d_ws;
    float* s_src = (float*)ws;                                    // R*N
    float* s_dst = s_src + (size_t)R * N;                         // R*N
    float* agg   = s_dst + (size_t)R * N;                         // R*N*128
    float* h     = agg + (size_t)R * N * 128;                     // N*128
    unsigned short* Wt1 = (unsigned short*)(h + (size_t)N * 128); // R*128*128
    unsigned short* Wt2 = Wt1 + (size_t)R * 128 * 128;            // R*128*128

    const int BT = 256;
    dim3 blk(BT);

    // Degrees + normalization factors (s_src/s_dst contiguous -> one norm pass)
    hipMemsetAsync(s_src, 0, sizeof(float) * 2 * (size_t)R * N, stream);
    rgcn_degrees<<<dim3((RE + BT - 1) / BT), blk, 0, stream>>>(src, dst, s_src, s_dst, RE, E, N);
    rgcn_norms<<<dim3((2 * R * N + BT - 1) / BT), blk, 0, stream>>>(s_src, 2 * R * N);

    // Pre-convert both layers' weights to bf16 [r][n][k]
    const int WTOT = R * 128 * 128;
    rgcn_wconv<<<dim3((WTOT + BT - 1) / BT), blk, 0, stream>>>(W1, Wt1, WTOT);
    rgcn_wconv<<<dim3((WTOT + BT - 1) / BT), blk, 0, stream>>>(W2, Wt2, WTOT);

    const int scat_blocks = (RE * 32 + BT - 1) / BT;
    const int gemm_blocks = (N + 15) / 16;

    // Layer 1
    hipMemsetAsync(agg, 0, sizeof(float) * (size_t)R * N * 128, stream);
    rgcn_scatter<<<dim3(scat_blocks), blk, 0, stream>>>(x, src, dst, s_src, agg, RE, E, N);
    rgcn_gemm<<<dim3(gemm_blocks), blk, 0, stream>>>(agg, s_dst, Wt1, b1, h, N, R, 1);

    // Layer 2
    hipMemsetAsync(agg, 0, sizeof(float) * (size_t)R * N * 128, stream);
    rgcn_scatter<<<dim3(scat_blocks), blk, 0, stream>>>(h, src, dst, s_src, agg, RE, E, N);
    rgcn_gemm<<<dim3(gemm_blocks), blk, 0, stream>>>(agg, s_dst, Wt2, b2, out, N, R, 0);
}
